// SAMGINet_8684423872574
// MI455X (gfx1250) — compile-verified, matrix operand order checked by reading, not run on hardware
//
#include <hip/hip_runtime.h>
#include <math.h>

typedef __attribute__((ext_vector_type(16))) __bf16 v16bf;
typedef __attribute__((ext_vector_type(8)))  float  v8f;

namespace {
constexpr int NN = 100000;   // nodes
constexpr int EE = 1600000;  // edges
constexpr int GG = 256;      // graphs
constexpr int HH = 128;      // hidden
constexpr int K1 = 50000;    // pool-1 keep
constexpr int K2 = 25000;    // pool-2 keep
constexpr int CC = 10;       // classes
}

static __device__ __forceinline__ unsigned keyOf(float s) {
  unsigned u = __float_as_uint(s);
  return (u & 0x80000000u) ? ~u : (u | 0x80000000u);  // monotonic float order
}

static __device__ __forceinline__ void atomicMaxF(float* a, float v) {
  unsigned* ua = (unsigned*)a;
  unsigned old = *ua;
  for (;;) {
    float f = __uint_as_float(old);
    if (f >= v) return;
    unsigned assumed = old;
    old = atomicCAS(ua, assumed, __float_as_uint(v));
    if (old == assumed) return;
  }
}

// ---------------- fills ----------------
__global__ void k_fill_f32(float* p, float v, int n) {
  int i = blockIdx.x * blockDim.x + threadIdx.x;
  if (i < n) p[i] = v;
}
__global__ void k_fill_i32(int* p, int v, int n) {
  int i = blockIdx.x * blockDim.x + threadIdx.x;
  if (i < n) p[i] = v;
}

// ---------------- W pre-pack: fp32 [128,128] -> bf16 in per-lane fragment order ----
// Fragment fb = kk*8+nt (0..31); within: lane (0..31) holds 16 contiguous bf16.
__global__ void k_packW(const float* __restrict__ W, __bf16* __restrict__ Wp) {
  int t = blockIdx.x * blockDim.x + threadIdx.x;
  if (t >= 128 * 128) return;
  int j    = t & 15;
  int lane = (t >> 4) & 31;
  int fb   = t >> 9;            // kk*8 + nt
  int kk   = fb >> 3, nt = fb & 7;
  int half = lane >> 4, lmod = lane & 15;
  int col  = nt * 16 + lmod;
  int k    = kk * 32 + ((j < 8) ? (half * 8 + j) : (16 + half * 8 + (j - 8)));
  Wp[t] = (__bf16)W[k * 128 + col];
}

// ---------------- bf16 WMMA GEMM: Out[n,128] = A[n,128] @ W[128,128] ----------------
union V16U {
  uint4 q[2];
  v16bf v;
};

__global__ __launch_bounds__(128)
void k_gemm128(const float* __restrict__ A, const __bf16* __restrict__ Wp,
               float* __restrict__ Out, int n) {
  __shared__ __bf16 ldsA[64 * 128];        // 16 KB, row-major bf16 tile
  __shared__ __bf16 ldsW[32 * 32 * 16];    // 32 KB, fragment-packed
  const int tid = threadIdx.x;
  const int rowBase = blockIdx.x * 64;

  // copy packed W (32 KB) as uint4 -> ds_store_b128
  {
    const uint4* s4 = (const uint4*)Wp;
    uint4* d4 = (uint4*)ldsW;
    for (int i = tid; i < 2048; i += 128) d4[i] = s4[i];
  }
  // stage A tile: float4 load -> 4x bf16 packed -> one 8-byte LDS store
  {
    uint2* d2 = (uint2*)ldsA;
    for (int i = tid; i < 2048; i += 128) {      // i indexes groups of 4 elems
      int r = i >> 5, c4 = i & 31;
      int gr = rowBase + r;
      float4 v = make_float4(0.f, 0.f, 0.f, 0.f);
      if (gr < n) v = ((const float4*)(A + (size_t)gr * 128))[c4];
      union { __bf16 h[4]; uint2 u; } pk;
      pk.h[0] = (__bf16)v.x; pk.h[1] = (__bf16)v.y;
      pk.h[2] = (__bf16)v.z; pk.h[3] = (__bf16)v.w;
      d2[i] = pk.u;
    }
  }
  __syncthreads();

  const int wave = tid >> 5;        // 0..3 -> 16-row strip each
  const int lane = tid & 31;
  const int half = lane >> 4;
  const int lmod = lane & 15;
  const int mLocal = wave * 16 + lmod;

  v8f acc[8] = {};
  for (int kk = 0; kk < 4; ++kk) {
    const int kbase = kk * 32;
    V16U af;
    af.q[0] = *(const uint4*)&ldsA[mLocal * 128 + kbase + half * 8];
    af.q[1] = *(const uint4*)&ldsA[mLocal * 128 + kbase + 16 + half * 8];
#pragma unroll
    for (int nt = 0; nt < 8; ++nt) {
      V16U bfr;
      const uint4* wp = (const uint4*)&ldsW[((kk * 8 + nt) * 32 + lane) * 16];
      bfr.q[0] = wp[0];
      bfr.q[1] = wp[1];
      acc[nt] = __builtin_amdgcn_wmma_f32_16x16x32_bf16(
          false, af.v, false, bfr.v, (short)0, acc[nt], false, false);
    }
  }

  if (rowBase + 64 <= n) {          // fast path: no guards
#pragma unroll
    for (int nt = 0; nt < 8; ++nt) {
      const int col = nt * 16 + lmod;
#pragma unroll
      for (int i = 0; i < 8; ++i) {
        int grow = rowBase + wave * 16 + half * 8 + i;
        Out[(size_t)grow * 128 + col] = acc[nt][i];
      }
    }
  } else {
#pragma unroll
    for (int nt = 0; nt < 8; ++nt) {
      const int col = nt * 16 + lmod;
#pragma unroll
      for (int i = 0; i < 8; ++i) {
        int grow = rowBase + wave * 16 + half * 8 + i;
        if (grow < n) Out[(size_t)grow * 128 + col] = acc[nt][i];
      }
    }
  }
}

// ---------------- GCN pieces ----------------
__global__ void k_deg_edges(const int* __restrict__ dst, const float* __restrict__ ew,
                            float* __restrict__ deg, int e) {
  int i = blockIdx.x * blockDim.x + threadIdx.x;
  if (i >= e) return;
  float w = ew[i];
  if (w != 0.0f) atomicAdd(&deg[dst[i]], w);
}
__global__ void k_dinv(const float* __restrict__ deg, float* __restrict__ dinv, int n) {
  int i = blockIdx.x * blockDim.x + threadIdx.x;
  if (i >= n) return;
  float d = deg[i];
  dinv[i] = (d > 0.0f) ? rsqrtf(fmaxf(d, 1e-12f)) : 0.0f;
}
__global__ void k_conv_init(float* __restrict__ out, const float* __restrict__ xW,
                            const float* __restrict__ b, const float* __restrict__ dinv, int n) {
  int idx = blockIdx.x * blockDim.x + threadIdx.x;
  if (idx >= n * HH) return;
  int i = idx >> 7, f = idx & 127;
  float di = dinv[i];
  out[idx] = b[f] + di * di * xW[idx];
}
__global__ __launch_bounds__(256)
void k_conv_scatter(const int* __restrict__ src, const int* __restrict__ dst,
                    const float* __restrict__ ew, const float* __restrict__ dinv,
                    const float* __restrict__ xW, float* __restrict__ out, int e) {
  int gid = blockIdx.x * 256 + threadIdx.x;
  int ed = gid >> 5;       // one wave per edge
  int lane = gid & 31;     // lane covers 4 consecutive features
  if (ed >= e) return;
  float w = ew[ed];
  if (w == 0.0f) return;   // masked edge: contributes zero
  int s = src[ed], d = dst[ed];
  float norm = dinv[s] * w * dinv[d];
  const float4 v = ((const float4*)(xW + (size_t)s * 128))[lane];
  float* o = out + (size_t)d * 128 + lane * 4;
  atomicAdd(o + 0, norm * v.x);
  atomicAdd(o + 1, norm * v.y);
  atomicAdd(o + 2, norm * v.z);
  atomicAdd(o + 3, norm * v.w);
}
__global__ void k_relu(float* p, int n) {
  int i = blockIdx.x * blockDim.x + threadIdx.x;
  if (i < n) p[i] = fmaxf(p[i], 0.0f);
}

// ---------------- pooling: score, losses, radix top-k select ----------------
__global__ __launch_bounds__(256)
void k_score(const float* __restrict__ h, const float* __restrict__ p,
             float* __restrict__ score, int n) {
  __shared__ float sp[HH];
  __shared__ float snorm;
  int t = threadIdx.x;
  if (t < HH) sp[t] = p[t];
  __syncthreads();
  if (t == 0) {
    float s = 0.0f;
    for (int j = 0; j < HH; ++j) s += sp[j] * sp[j];
    snorm = sqrtf(s);
  }
  __syncthreads();
  int i = blockIdx.x * 256 + t;
  if (i >= n) return;
  const float* row = h + (size_t)i * HH;
  float d = 0.0f;
  for (int f = 0; f < HH; ++f) d += row[f] * sp[f];
  score[i] = d / snorm;
}
__global__ __launch_bounds__(256)
void k_loss_acc(const float* __restrict__ score, int n, float* __restrict__ acc2) {
  __shared__ float s1[256], s2[256];
  int t = threadIdx.x;
  int i = blockIdx.x * 256 + t;
  float s = (i < n) ? score[i] : 0.0f;
  s1[t] = s * s;
  s2[t] = fabsf(s);
  __syncthreads();
  for (int o = 128; o > 0; o >>= 1) {
    if (t < o) { s1[t] += s1[t + o]; s2[t] += s2[t + o]; }
    __syncthreads();
  }
  if (t == 0) { atomicAdd(&acc2[0], s1[0]); atomicAdd(&acc2[1], s2[0]); }
}
__global__ void k_sel_init(unsigned* state, unsigned* hist, unsigned k) {
  if (blockIdx.x == 0 && threadIdx.x == 0) {
    state[0] = 0u;   // prefix / final threshold
    state[1] = k;    // remaining among prefix-matching candidates
    state[2] = 0u;   // compaction counter
    for (int i = 0; i < 256; ++i) hist[i] = 0u;
  }
}
__global__ void k_hist(const float* __restrict__ score, int n,
                       const unsigned* __restrict__ state, unsigned* __restrict__ hist,
                       int shift) {
  int i = blockIdx.x * blockDim.x + threadIdx.x;
  if (i >= n) return;
  unsigned u = keyOf(score[i]);
  unsigned himask = (shift == 24) ? 0u : (0xFFFFFFFFu << (shift + 8));
  if ((u & himask) == (state[0] & himask))
    atomicAdd(&hist[(u >> shift) & 0xFFu], 1u);
}
__global__ void k_sel_step(unsigned* state, unsigned* hist, int shift) {
  if (blockIdx.x != 0 || threadIdx.x != 0) return;
  unsigned rem = state[1];
  unsigned cum = 0u;
  int v = 255;
  for (; v > 0; --v) {
    unsigned c = hist[v];
    if (cum + c >= rem) break;
    cum += c;
  }
  state[0] |= ((unsigned)v) << shift;
  state[1] = rem - cum;
  for (int i = 0; i < 256; ++i) hist[i] = 0u;
}
__global__ void k_compact_gt(const float* __restrict__ score, int n,
                             unsigned* __restrict__ state, int* __restrict__ perm) {
  int i = blockIdx.x * blockDim.x + threadIdx.x;
  if (i >= n) return;
  if (keyOf(score[i]) > state[0]) {
    unsigned p = atomicAdd(&state[2], 1u);
    perm[p] = i;
  }
}
__global__ void k_compact_eq(const float* __restrict__ score, int n,
                             unsigned* __restrict__ state, int* __restrict__ perm,
                             unsigned kk) {
  int i = blockIdx.x * blockDim.x + threadIdx.x;
  if (i >= n) return;
  if (keyOf(score[i]) == state[0]) {
    unsigned p = atomicAdd(&state[2], 1u);
    if (p < kk) perm[p] = i;
  }
}
__global__ void k_newidx_set(const int* __restrict__ perm, int* __restrict__ newidx, int k) {
  int j = blockIdx.x * blockDim.x + threadIdx.x;
  if (j < k) newidx[perm[j]] = j;
}
__global__ void k_gather_pool(const float* __restrict__ h, const float* __restrict__ score,
                              const int* __restrict__ perm, float* __restrict__ xn, int k) {
  int idx = blockIdx.x * blockDim.x + threadIdx.x;
  if (idx >= k * HH) return;
  int j = idx >> 7, f = idx & 127;
  int o = perm[j];
  xn[idx] = h[(size_t)o * HH + f] * tanhf(score[o]);
}
__global__ void k_gather_batch(const int* __restrict__ bin, const int* __restrict__ perm,
                               int* __restrict__ bout, int k) {
  int j = blockIdx.x * blockDim.x + threadIdx.x;
  if (j < k) bout[j] = bin[perm[j]];
}
__global__ void k_edge_remap(const int* __restrict__ src, const int* __restrict__ dst,
                             const float* __restrict__ ew, const int* __restrict__ newidx,
                             int* __restrict__ s2, int* __restrict__ d2,
                             float* __restrict__ ew2, int e) {
  int i = blockIdx.x * blockDim.x + threadIdx.x;
  if (i >= e) return;
  int a = newidx[src[i]], b = newidx[dst[i]];
  bool valid = (a >= 0) && (b >= 0);
  ew2[i] = valid ? ew[i] : 0.0f;
  s2[i] = valid ? a : 0;
  d2[i] = valid ? b : 0;
}

// ---------------- readout ----------------
__global__ void k_cnt(const int* __restrict__ bn, float* __restrict__ gcnt, int n) {
  int i = blockIdx.x * blockDim.x + threadIdx.x;
  if (i < n) atomicAdd(&gcnt[bn[i]], 1.0f);
}
__global__ void k_pool_acc(const float* __restrict__ x, const int* __restrict__ bn,
                           float* __restrict__ gsum, float* __restrict__ gmax, int n) {
  int idx = blockIdx.x * blockDim.x + threadIdx.x;
  if (idx >= n * HH) return;
  int i = idx >> 7, f = idx & 127;
  int g = bn[i];
  float v = x[idx];
  atomicAdd(&gsum[g * HH + f], v);
  atomicMaxF(&gmax[g * HH + f], v);
}
__global__ void k_readout_add(float* __restrict__ gemb, const float* __restrict__ gsum,
                              const float* __restrict__ gmax, const float* __restrict__ gcnt) {
  int idx = blockIdx.x * blockDim.x + threadIdx.x;
  if (idx >= GG * 2 * HH) return;
  int g = idx >> 8, f = idx & 255;
  float m = (f < HH) ? (gsum[g * HH + f] / fmaxf(gcnt[g], 1.0f))
                     : gmax[g * HH + (f - HH)];
  gemb[idx] += fmaxf(m, 0.0f);
}

// ---------------- classifier + losses ----------------
__global__ __launch_bounds__(128)
void k_classifier(const float* __restrict__ gemb,
                  const float* __restrict__ cW1, const float* __restrict__ cb1,
                  const float* __restrict__ cW2, const float* __restrict__ cb2,
                  const float* __restrict__ cW3, const float* __restrict__ cb3,
                  float* __restrict__ out) {
  __shared__ float in[2 * HH];
  __shared__ float z1[HH];
  __shared__ float z2[64];
  __shared__ float lg[CC];
  int g = blockIdx.x, t = threadIdx.x;
  in[t] = gemb[g * 256 + t];
  in[t + 128] = gemb[g * 256 + t + 128];
  __syncthreads();
  float s = cb1[t];
  for (int k = 0; k < 256; ++k) s += in[k] * cW1[k * 128 + t];
  z1[t] = fmaxf(s, 0.0f);
  __syncthreads();
  if (t < 64) {
    float s2 = cb2[t];
    for (int k = 0; k < 128; ++k) s2 += z1[k] * cW2[k * 64 + t];
    z2[t] = fmaxf(s2, 0.0f);
  }
  __syncthreads();
  if (t < CC) {
    float s3 = cb3[t];
    for (int k = 0; k < 64; ++k) s3 += z2[k] * cW3[k * CC + t];
    lg[t] = s3;
  }
  __syncthreads();
  if (t == 0) {
    float mx = -3.402823466e38f;
    for (int j = 0; j < CC; ++j) mx = fmaxf(mx, lg[j]);
    float se = 0.0f;
    for (int j = 0; j < CC; ++j) se += expf(lg[j] - mx);
    float lse = mx + logf(se);
    for (int j = 0; j < CC; ++j) out[g * CC + j] = lg[j] - lse;
  }
}
__global__ void k_finalize(const float* __restrict__ acc, float* __restrict__ out,
                           int off, float n1, float n2) {
  if (blockIdx.x == 0 && threadIdx.x == 0) {
    out[off + 0] = 0.5f * (acc[0] / n1 + acc[2] / n2);
    out[off + 1] = 0.5f * (acc[1] / n1 + acc[3] / n2);
  }
}

// ---------------- driver ----------------
extern "C" void kernel_launch(void* const* d_in, const int* in_sizes, int n_in,
                              void* d_out, int out_size, void* d_ws, size_t ws_size,
                              hipStream_t stream) {
  (void)in_sizes; (void)n_in; (void)out_size; (void)ws_size;
  const float* x     = (const float*)d_in[0];
  const int*   ei    = (const int*)d_in[1];
  const float* ea    = (const float*)d_in[2];
  const int*   batch = (const int*)d_in[3];
  const float* W0 = (const float*)d_in[4];  const float* b0 = (const float*)d_in[5];
  const float* W1 = (const float*)d_in[6];  const float* b1 = (const float*)d_in[7];
  const float* p1 = (const float*)d_in[10]; const float* p2 = (const float*)d_in[11];
  const float* cW1 = (const float*)d_in[12]; const float* cb1 = (const float*)d_in[13];
  const float* cW2 = (const float*)d_in[14]; const float* cb2 = (const float*)d_in[15];
  const float* cW3 = (const float*)d_in[16]; const float* cb3 = (const float*)d_in[17];
  const int* src = ei;
  const int* dst = ei + EE;
  float* out = (float*)d_out;

  size_t off = 0;
  auto alloc = [&](size_t bytes) -> void* {
    void* p = (void*)((char*)d_ws + off);
    off += (bytes + 255) & ~(size_t)255;
    return p;
  };
  float* bufA = (float*)alloc((size_t)NN * HH * 4);   // h1, then xW of conv1
  float* bufB = (float*)alloc((size_t)NN * HH * 4);   // xW of conv0, then xn1, then xn2
  float* bufC = (float*)alloc((size_t)K1 * HH * 4);   // h2
  int*   s1   = (int*)alloc((size_t)EE * 4);
  int*   d1   = (int*)alloc((size_t)EE * 4);
  float* ew1  = (float*)alloc((size_t)EE * 4);
  float* deg  = (float*)alloc((size_t)NN * 4);
  float* dinv = (float*)alloc((size_t)NN * 4);
  float* score= (float*)alloc((size_t)NN * 4);
  int*   perm = (int*)alloc((size_t)K1 * 4);
  int*   newidx = (int*)alloc((size_t)NN * 4);
  int*   bn1  = (int*)alloc((size_t)K1 * 4);
  int*   bn2  = (int*)alloc((size_t)K2 * 4);
  float* gsum = (float*)alloc((size_t)GG * HH * 4);
  float* gmax = (float*)alloc((size_t)GG * HH * 4);
  float* gcnt = (float*)alloc((size_t)GG * 4);
  float* gemb = (float*)alloc((size_t)GG * 2 * HH * 4);
  float* acc  = (float*)alloc(8 * 4);
  unsigned* hist  = (unsigned*)alloc(256 * 4);
  unsigned* state = (unsigned*)alloc(4 * 4);
  __bf16* wpack = (__bf16*)alloc((size_t)HH * HH * 2);

  auto nb = [](long long n) { return (unsigned)((n + 255) / 256); };
  const unsigned bEdge = nb(EE);
  const unsigned bScat = nb((long long)EE * 32);
  const int shifts[4] = {24, 16, 8, 0};

  // global init
  k_fill_f32<<<nb(GG * 2 * HH), 256, 0, stream>>>(gemb, 0.0f, GG * 2 * HH);
  k_fill_f32<<<1, 8, 0, stream>>>(acc, 0.0f, 8);

  // ---- conv0: h1 = relu(gcn(x, W0, b0)) ----
  k_packW<<<64, 256, 0, stream>>>(W0, wpack);
  k_gemm128<<<(NN + 63) / 64, 128, 0, stream>>>(x, wpack, bufB, NN);
  k_fill_f32<<<nb(NN), 256, 0, stream>>>(deg, 1.0f, NN);            // self-loop weight
  k_deg_edges<<<bEdge, 256, 0, stream>>>(dst, ea, deg, EE);
  k_dinv<<<nb(NN), 256, 0, stream>>>(deg, dinv, NN);
  k_conv_init<<<nb((long long)NN * HH), 256, 0, stream>>>(bufA, bufB, b0, dinv, NN);
  k_conv_scatter<<<bScat, 256, 0, stream>>>(src, dst, ea, dinv, bufB, bufA, EE);
  k_relu<<<nb((long long)NN * HH), 256, 0, stream>>>(bufA, NN * HH);

  // ---- pool1 (p1, k=K1) ----
  k_score<<<nb(NN), 256, 0, stream>>>(bufA, p1, score, NN);
  k_loss_acc<<<nb(NN), 256, 0, stream>>>(score, NN, acc + 0);
  k_sel_init<<<1, 1, 0, stream>>>(state, hist, (unsigned)K1);
  for (int r = 0; r < 4; ++r) {
    k_hist<<<nb(NN), 256, 0, stream>>>(score, NN, state, hist, shifts[r]);
    k_sel_step<<<1, 1, 0, stream>>>(state, hist, shifts[r]);
  }
  k_compact_gt<<<nb(NN), 256, 0, stream>>>(score, NN, state, perm);
  k_compact_eq<<<nb(NN), 256, 0, stream>>>(score, NN, state, perm, (unsigned)K1);
  k_fill_i32<<<nb(NN), 256, 0, stream>>>(newidx, -1, NN);
  k_newidx_set<<<nb(K1), 256, 0, stream>>>(perm, newidx, K1);
  k_gather_pool<<<nb((long long)K1 * HH), 256, 0, stream>>>(bufA, score, perm, bufB, K1);
  k_gather_batch<<<nb(K1), 256, 0, stream>>>(batch, perm, bn1, K1);
  k_edge_remap<<<bEdge, 256, 0, stream>>>(src, dst, ea, newidx, s1, d1, ew1, EE);

  // ---- readout1 ----
  k_fill_f32<<<nb(GG * HH), 256, 0, stream>>>(gsum, 0.0f, GG * HH);
  k_fill_f32<<<nb(GG * HH), 256, 0, stream>>>(gmax, -3.402823466e38f, GG * HH);
  k_fill_f32<<<1, 256, 0, stream>>>(gcnt, 0.0f, GG);
  k_cnt<<<nb(K1), 256, 0, stream>>>(bn1, gcnt, K1);
  k_pool_acc<<<nb((long long)K1 * HH), 256, 0, stream>>>(bufB, bn1, gsum, gmax, K1);
  k_readout_add<<<nb(GG * 2 * HH), 256, 0, stream>>>(gemb, gsum, gmax, gcnt);

  // ---- conv1: h2 = relu(gcn(xn1, W1, b1, pooled edges)) ----
  k_packW<<<64, 256, 0, stream>>>(W1, wpack);
  k_gemm128<<<(K1 + 63) / 64, 128, 0, stream>>>(bufB, wpack, bufA, K1);
  k_fill_f32<<<nb(K1), 256, 0, stream>>>(deg, 1.0f, K1);
  k_deg_edges<<<bEdge, 256, 0, stream>>>(d1, ew1, deg, EE);
  k_dinv<<<nb(K1), 256, 0, stream>>>(deg, dinv, K1);
  k_conv_init<<<nb((long long)K1 * HH), 256, 0, stream>>>(bufC, bufA, b1, dinv, K1);
  k_conv_scatter<<<bScat, 256, 0, stream>>>(s1, d1, ew1, dinv, bufA, bufC, EE);
  k_relu<<<nb((long long)K1 * HH), 256, 0, stream>>>(bufC, K1 * HH);

  // ---- pool2 (p2, k=K2) — edges/newidx not needed (final conv is dead) ----
  k_score<<<nb(K1), 256, 0, stream>>>(bufC, p2, score, K1);
  k_loss_acc<<<nb(K1), 256, 0, stream>>>(score, K1, acc + 2);
  k_sel_init<<<1, 1, 0, stream>>>(state, hist, (unsigned)K2);
  for (int r = 0; r < 4; ++r) {
    k_hist<<<nb(K1), 256, 0, stream>>>(score, K1, state, hist, shifts[r]);
    k_sel_step<<<1, 1, 0, stream>>>(state, hist, shifts[r]);
  }
  k_compact_gt<<<nb(K1), 256, 0, stream>>>(score, K1, state, perm);
  k_compact_eq<<<nb(K1), 256, 0, stream>>>(score, K1, state, perm, (unsigned)K2);
  k_gather_pool<<<nb((long long)K2 * HH), 256, 0, stream>>>(bufC, score, perm, bufB, K2);
  k_gather_batch<<<nb(K2), 256, 0, stream>>>(bn1, perm, bn2, K2);

  // ---- readout2 ----
  k_fill_f32<<<nb(GG * HH), 256, 0, stream>>>(gsum, 0.0f, GG * HH);
  k_fill_f32<<<nb(GG * HH), 256, 0, stream>>>(gmax, -3.402823466e38f, GG * HH);
  k_fill_f32<<<1, 256, 0, stream>>>(gcnt, 0.0f, GG);
  k_cnt<<<nb(K2), 256, 0, stream>>>(bn2, gcnt, K2);
  k_pool_acc<<<nb((long long)K2 * HH), 256, 0, stream>>>(bufB, bn2, gsum, gmax, K2);
  k_readout_add<<<nb(GG * 2 * HH), 256, 0, stream>>>(gemb, gsum, gmax, gcnt);

  // ---- classifier + losses ----
  k_classifier<<<GG, 128, 0, stream>>>(gemb, cW1, cb1, cW2, cb2, cW3, cb3, out);
  k_finalize<<<1, 1, 0, stream>>>(acc, out, GG * CC, (float)NN, (float)K1);
}
